// ECCMLayer_42399917146740
// MI455X (gfx1250) — matrix operand order repaired and checked by reading, not compiled, not run on hardware
//
#include <hip/hip_runtime.h>
#include <hip/hip_bf16.h>
#include <math.h>

// ---------------- problem dims (compile-time) ----------------
#define BATCH   64
#define LSEQ    192
#define DMODEL  512
#define DINNER  512
#define DSTATE  256
#define DTRANK  32
#define DCONV   4
#define XDBL    (DTRANK + 2*DSTATE)   // 544
#define LOG2E   1.44269504088896340736f

typedef __bf16 bf16_t;
typedef __attribute__((ext_vector_type(16))) __bf16 v16bf;
typedef __attribute__((ext_vector_type(8)))  __bf16 v8bf;
typedef __attribute__((ext_vector_type(8)))  float  v8f;

// ---------------- helpers ----------------
__global__ void f32_to_bf16_kernel(const float* __restrict__ in,
                                   bf16_t* __restrict__ out, int n) {
    int i = blockIdx.x * blockDim.x + threadIdx.x;
    if (i < n) out[i] = (bf16_t)in[i];
}

// Aneg2[i] = -exp(A_log[i]) * log2(e)   (pre-scaled for native v_exp_f32)
__global__ void aneg_kernel(const float* __restrict__ A_log,
                            float* __restrict__ Aneg2, int n) {
    int i = blockIdx.x * blockDim.x + threadIdx.x;
    if (i < n) Aneg2[i] = -__expf(A_log[i]) * LOG2E;
}

// ---------------- bf16 WMMA GEMM: C[m,n] = sum_k A[m,k]*BT[n,k] ----------------
// Register-blocked 1x4 in N: one wave computes a 16x64 tile of C,
// reusing each A fragment across 4 v_wmma ops (4x arithmetic intensity).
// A:  row-major  [Mtot x K]  (bf16)
// BT: n-major    [Ntot x K]  (bf16)  (B transposed, K contiguous)
// C:  fp32. storeT==0: C[m*ldc + n]; storeT==1: C[n*ldc + m]
// grid = (Ntot/64, Mtot/16, BATCH), block = 32 (one wave)
__global__ __launch_bounds__(32)
void gemm_bf16_wmma(const bf16_t* __restrict__ A,
                    const bf16_t* __restrict__ BT,
                    float* __restrict__ C,
                    int K, long long strideB, long long strideC,
                    int ldc, int storeT) {
    const int n0 = blockIdx.x * 64;
    const int m0 = blockIdx.y * 16;
    const int b  = blockIdx.z;
    const bf16_t* Bb = BT + (long long)b * strideB;
    float*        Cb = C  + (long long)b * strideC;

    const int lane = threadIdx.x;           // 0..31, wave32
    const int half = lane >> 4;             // 0 or 1
    const int sub  = lane & 15;

    // A layout: lane -> row M=sub; 16 contiguous K starting at half*16
    const int mrow   = m0 + sub;
    const int kbaseA = half * 16;
    // B layout: lane -> col N=sub (+16j); 16 contiguous K starting at half*16
    const int kbaseB = half * 16;

    v8f acc0 = {}, acc1 = {}, acc2 = {}, acc3 = {};
    const bf16_t* bcol = Bb + (long long)(n0 + sub) * K + kbaseB;

    for (int k = 0; k < K; k += 32) {
        const bf16_t* ap = A + (long long)mrow * K + k + kbaseA;
        v8bf alo = *(const v8bf*)(ap);
        v8bf ahi = *(const v8bf*)(ap + 8);
        v16bf av;
#pragma unroll
        for (int i = 0; i < 8; ++i) { av[i] = alo[i]; av[8 + i] = ahi[i]; }

        v16bf bv0 = *(const v16bf*)(bcol + 0 * 16 * K + k);
        v16bf bv1 = *(const v16bf*)(bcol + 1 * 16 * K + k);
        v16bf bv2 = *(const v16bf*)(bcol + 2 * 16 * K + k);
        v16bf bv3 = *(const v16bf*)(bcol + 3 * 16 * K + k);

        acc0 = __builtin_amdgcn_wmma_f32_16x16x32_bf16(false, av, false, bv0, (short)0, acc0, false, false);
        acc1 = __builtin_amdgcn_wmma_f32_16x16x32_bf16(false, av, false, bv1, (short)0, acc1, false, false);
        acc2 = __builtin_amdgcn_wmma_f32_16x16x32_bf16(false, av, false, bv2, (short)0, acc2, false, false);
        acc3 = __builtin_amdgcn_wmma_f32_16x16x32_bf16(false, av, false, bv3, (short)0, acc3, false, false);
    }

    // C/D layout: VGPR r -> M = r + 8*half, N = sub (+16j)
    const int roff = half * 8;
    v8f accs[4] = {acc0, acc1, acc2, acc3};
#pragma unroll
    for (int j = 0; j < 4; ++j) {
        const int ncol = n0 + 16 * j + sub;
        if (!storeT) {
#pragma unroll
            for (int r = 0; r < 8; ++r)
                Cb[(long long)(m0 + roff + r) * ldc + ncol] = accs[j][r];
        } else {
            float* p = Cb + (long long)ncol * ldc + (m0 + roff);
#pragma unroll
            for (int r = 0; r < 8; ++r) p[r] = accs[j][r];
        }
    }
}

// ---------------- depthwise causal conv(k=4) + bias + SiLU ----------------
// reads xz[b, d, l] (x half); writes xconv_f32[b,d,l] and xconvT_bf16[b,l,d]
__global__ void conv_silu_kernel(const float* __restrict__ xz,
                                 const float* __restrict__ conv_w,
                                 const float* __restrict__ conv_b,
                                 float* __restrict__ xc,
                                 bf16_t* __restrict__ xcT) {
    int idx = blockIdx.x * blockDim.x + threadIdx.x;
    if (idx >= BATCH * DINNER * LSEQ) return;
    int l = idx % LSEQ;
    int d = (idx / LSEQ) % DINNER;
    int b = idx / (LSEQ * DINNER);

    const float* row = xz + ((long long)b * (2 * DINNER) + d) * LSEQ;
    float v = conv_b[d];
#pragma unroll
    for (int j = 0; j < DCONV; ++j) {
        int ll = l + j - (DCONV - 1);
        if (ll >= 0) v += conv_w[d * DCONV + j] * row[ll];
    }
    float s = v * (1.0f / (1.0f + __expf(-v)));   // SiLU
    xc[idx] = s;
    xcT[((long long)b * LSEQ + l) * DINNER + d] = (bf16_t)s;
}

// ---------------- delta = softplus(dt_proj @ dt_low + dt_bias) ----------------
__global__ void delta_kernel(const float* __restrict__ x_dbl,
                             const float* __restrict__ dt_proj_w,
                             const float* __restrict__ dt_bias,
                             float* __restrict__ delta) {
    int idx = blockIdx.x * blockDim.x + threadIdx.x;
    if (idx >= BATCH * DINNER * LSEQ) return;
    int l = idx % LSEQ;
    int d = (idx / LSEQ) % DINNER;
    int b = idx / (LSEQ * DINNER);

    const float* xd = x_dbl + (long long)b * XDBL * LSEQ;  // rows 0..DTRANK-1 = dt_low
    float acc = dt_bias[d];
#pragma unroll
    for (int r = 0; r < DTRANK; ++r)
        acc += dt_proj_w[d * DTRANK + r] * xd[r * LSEQ + l];
    float sp = (acc > 20.0f) ? acc : log1pf(__expf(acc));  // stable softplus
    delta[idx] = sp;
}

// ---------------- selective scan ----------------
// block = 256 threads = 16 d-values x 16 state-slices (16 n each, in VGPRs)
// grid  = BATCH * (DINNER/16)
__global__ __launch_bounds__(256)
void scan_kernel(const float* __restrict__ x_dbl,
                 const float* __restrict__ delta,
                 const float* __restrict__ xc,
                 const float* __restrict__ Aneg2,
                 const unsigned char* __restrict__ pc_mask,
                 float* __restrict__ yscan) {
    const int b    = blockIdx.x / (DINNER / 16);
    const int dblk = blockIdx.x % (DINNER / 16);
    const int tid  = threadIdx.x;
    const int dloc = tid >> 4;          // 0..15
    const int ns   = tid & 15;          // state slice
    const int d    = dblk * 16 + dloc;
    const int n0   = ns * 16;

    __shared__ float Bs[DSTATE];
    __shared__ float Cs[DSTATE];

    const float* Bbase = x_dbl + (long long)b * XDBL * LSEQ + (long long)DTRANK * LSEQ;
    const float* Cbase = x_dbl + (long long)b * XDBL * LSEQ + (long long)(DTRANK + DSTATE) * LSEQ;
    const float* drow  = delta + ((long long)b * DINNER + d) * LSEQ;
    const float* urow  = xc    + ((long long)b * DINNER + d) * LSEQ;
    float*       yrow  = yscan + ((long long)b * (2 * DINNER) + d) * LSEQ; // overlays xz x-half

    float h[16];
    float an[16];                       // -exp(A_log)*log2(e), time-invariant
#pragma unroll
    for (int i = 0; i < 16; ++i) {
        h[i]  = 0.0f;
        an[i] = Aneg2[d * DSTATE + n0 + i];
    }

    for (int t = 0; t < LSEQ; ++t) {
        // stage masked B_t, C_t into LDS (one element per thread)
        {
            int n = tid;
            bool msk = (n < LSEQ) ? (t < 64 && pc_mask[n * 64 + t] != 0)
                                  : (t >= 64);
            Bs[n] = msk ? 0.0f : Bbase[(long long)n * LSEQ + t];
            Cs[n] = msk ? 0.0f : Cbase[(long long)n * LSEQ + t];
        }
        __syncthreads();

        float dlt = drow[t];
        float du  = dlt * urow[t];
        float acc = 0.0f;
#pragma unroll
        for (int i = 0; i < 16; ++i) {
            float ex = __builtin_amdgcn_exp2f(dlt * an[i]);  // native v_exp_f32
            h[i] = h[i] * ex + du * Bs[n0 + i];
            acc += h[i] * Cs[n0 + i];
        }
        // reduce over the 16 state-slice lanes (half-wave on wave32)
#pragma unroll
        for (int off = 8; off >= 1; off >>= 1)
            acc += __shfl_xor(acc, off, 16);
        if (ns == 0) yrow[t] = acc;
        __syncthreads();
    }
}

// ---------------- y = (yscan + D*x) * silu(z), mask, -> bf16 (b,l,d) ----------------
__global__ void gate_mask_kernel(const float* __restrict__ xz,  // yscan in x-half, z in z-half
                                 const float* __restrict__ xc,
                                 const float* __restrict__ D_param,
                                 const unsigned char* __restrict__ pc_mask,
                                 bf16_t* __restrict__ yT) {
    int idx = blockIdx.x * blockDim.x + threadIdx.x;
    if (idx >= BATCH * DINNER * LSEQ) return;
    int l = idx % LSEQ;
    int d = (idx / LSEQ) % DINNER;
    int b = idx / (LSEQ * DINNER);

    const float* base = xz + (long long)b * (2 * DINNER) * LSEQ;
    float y = base[(long long)d * LSEQ + l];                      // yscan (overlaid)
    float z = base[(long long)(DINNER + d) * LSEQ + l];           // z half
    y += D_param[d] * xc[idx];
    y *= z * (1.0f / (1.0f + __expf(-z)));
    bool msk = (d < LSEQ) ? (l < 64 && pc_mask[d * 64 + l] != 0) : (l >= 64);
    if (msk) y = 0.0f;
    yT[((long long)b * LSEQ + l) * DINNER + d] = (bf16_t)y;
}

// ---------------- launch ----------------
extern "C" void kernel_launch(void* const* d_in, const int* in_sizes, int n_in,
                              void* d_out, int out_size, void* d_ws, size_t ws_size,
                              hipStream_t stream) {
    const float* hidden   = (const float*)d_in[0];
    const unsigned char* pc_mask = (const unsigned char*)d_in[1]; // jax bool, 1 byte
    const float* in_proj  = (const float*)d_in[2];
    const float* conv_w   = (const float*)d_in[3];
    const float* conv_b   = (const float*)d_in[4];
    const float* x_proj   = (const float*)d_in[5];
    const float* dt_proj  = (const float*)d_in[6];
    const float* dt_bias  = (const float*)d_in[7];
    const float* A_log    = (const float*)d_in[8];
    const float* D_param  = (const float*)d_in[9];
    const float* out_proj = (const float*)d_in[10];
    float* out = (float*)d_out;

    // workspace carving (all sizes 256B-aligned)
    char* ws = (char*)d_ws;
    bf16_t* hid_bf   = (bf16_t*)ws;                   ws += (size_t)BATCH*LSEQ*DMODEL*2;
    bf16_t* w_in_bf  = (bf16_t*)ws;                   ws += (size_t)2*DINNER*DMODEL*2;
    bf16_t* w_xp_bf  = (bf16_t*)ws;                   ws += (size_t)XDBL*DINNER*2;
    bf16_t* w_out_bf = (bf16_t*)ws;                   ws += (size_t)DMODEL*DINNER*2;
    float*  xz       = (float*)ws;                    ws += (size_t)BATCH*2*DINNER*LSEQ*4;
    float*  xc       = (float*)ws;                    ws += (size_t)BATCH*DINNER*LSEQ*4;
    bf16_t* xcT      = (bf16_t*)ws;                   ws += (size_t)BATCH*LSEQ*DINNER*2;
    float*  x_dbl    = (float*)ws;                    ws += (size_t)BATCH*XDBL*LSEQ*4;
    float*  delta    = (float*)ws;                    ws += (size_t)BATCH*DINNER*LSEQ*4;
    float*  Aneg2    = (float*)ws;                    ws += (size_t)DINNER*DSTATE*4;
    bf16_t* yT       = (bf16_t*)ws;                   ws += (size_t)BATCH*LSEQ*DINNER*2;

    const int TB = 256;
    // bf16 conversions of weights + activations
    { int n = 2*DINNER*DMODEL; f32_to_bf16_kernel<<<(n+TB-1)/TB, TB, 0, stream>>>(in_proj,  w_in_bf,  n); }
    { int n = XDBL*DINNER;     f32_to_bf16_kernel<<<(n+TB-1)/TB, TB, 0, stream>>>(x_proj,   w_xp_bf,  n); }
    { int n = DMODEL*DINNER;   f32_to_bf16_kernel<<<(n+TB-1)/TB, TB, 0, stream>>>(out_proj, w_out_bf, n); }
    { int n = BATCH*LSEQ*DMODEL; f32_to_bf16_kernel<<<(n+TB-1)/TB, TB, 0, stream>>>(hidden, hid_bf, n); }
    { int n = DINNER*DSTATE;   aneg_kernel<<<(n+TB-1)/TB, TB, 0, stream>>>(A_log, Aneg2, n); }

    // GEMM1: xz[b,e,l] = in_proj @ hidden[b]^T
    {
        dim3 g(LSEQ/64, (2*DINNER)/16, BATCH);
        gemm_bf16_wmma<<<g, 32, 0, stream>>>(w_in_bf, hid_bf, xz, DMODEL,
            (long long)LSEQ*DMODEL, (long long)2*DINNER*LSEQ, LSEQ, 0);
    }
    // conv + SiLU
    { int n = BATCH*DINNER*LSEQ;
      conv_silu_kernel<<<(n+TB-1)/TB, TB, 0, stream>>>(xz, conv_w, conv_b, xc, xcT); }
    // GEMM2: x_dbl[b,o,l] = x_proj @ xconv[b]^T
    {
        dim3 g(LSEQ/64, XDBL/16, BATCH);
        gemm_bf16_wmma<<<g, 32, 0, stream>>>(w_xp_bf, xcT, x_dbl, DINNER,
            (long long)LSEQ*DINNER, (long long)XDBL*LSEQ, LSEQ, 0);
    }
    // delta
    { int n = BATCH*DINNER*LSEQ;
      delta_kernel<<<(n+TB-1)/TB, TB, 0, stream>>>(x_dbl, dt_proj, dt_bias, delta); }
    // selective scan (writes y into the consumed x-half of xz)
    {
        int blocks = BATCH * (DINNER/16);
        scan_kernel<<<blocks, 256, 0, stream>>>(x_dbl, delta, xc, Aneg2, pc_mask, xz);
    }
    // gate + mask -> yT (bf16, n-major)
    { int n = BATCH*DINNER*LSEQ;
      gate_mask_kernel<<<(n+TB-1)/TB, TB, 0, stream>>>(xz, xc, D_param, pc_mask, yT); }
    // GEMM3: out[b,l,o] = out_proj @ y[b]^T  (transposed store)
    {
        dim3 g(LSEQ/64, DMODEL/16, BATCH);
        gemm_bf16_wmma<<<g, 32, 0, stream>>>(w_out_bf, yT, out, DINNER,
            (long long)LSEQ*DINNER, (long long)LSEQ*DMODEL, DMODEL, 1);
    }
}
